// LoonBottleneck_58162447123326
// MI455X (gfx1250) — matrix operand order repaired and checked by reading, not compiled, not run on hardware
//
#include <hip/hip_runtime.h>
#include <hip/hip_bf16.h>
#include <stdint.h>

// ---------------------------------------------------------------------------
// CDNA5 / gfx1250 implementation of the offset-former point refiner.
// - KNN: LDS-tiled scan using async global->LDS (ASYNCcnt) + register top-K.
// - All dense linear layers: v_wmma_f32_16x16x32_f16 (fp16 in, fp32 acc),
//   fragments loaded as b128 vectors (A zero-padded in memory, B transposed).
// - Neighbor pipeline chunked (8192 pts) so fp16 intermediates live in L2.
// Input order assumes jax pytree (sorted-key) flattening of the inputs dict.
// ---------------------------------------------------------------------------

typedef __attribute__((ext_vector_type(16))) _Float16 v16h;
typedef __attribute__((ext_vector_type(8)))  _Float16 v8h;
typedef __attribute__((ext_vector_type(8)))  float    v8f;
typedef _Float16 h16;

#define NPTS 16384
#define NB   2
#define KNN  32
#define CHP  8192            // points per attention chunk

// ---------------------------------------------------------------------------
// WMMA GEMM: out[M,N] = act(A[M,Kpad] @ Bt[N,Kpad]^T + bias)
//  - A fp16 row-major, row stride lda (halfs); memory beyond logical K must be
//    readable & finite (we zero-pad buffers / zero-pad B rows instead of guards).
//  - Bt is the weight packed TRANSPOSED: Bt[n*Kpad + k], zero for k >= Kin.
//  - 8 waves per block, one 16x16 output tile per wave.
// ---------------------------------------------------------------------------
template<int ACT>
__launch_bounds__(256)
__global__ void gemm_wmma(const h16* __restrict__ A, int Kpad, int lda,
                          const h16* __restrict__ Bt, int N, int ntilesTot,
                          const float* __restrict__ bias,
                          h16* __restrict__ outH, float* __restrict__ outF)
{
  const int tile = blockIdx.x * 8 + (threadIdx.x >> 5);
  if (tile >= ntilesTot) return;               // wave-uniform exit
  const int ntN = N >> 4;
  const int tm = tile / ntN, tn = tile % ntN;
  const int lane = threadIdx.x & 31;
  const int sel  = lane >> 4;                  // half-wave selector
  const int l15  = lane & 15;
  v8f acc = {};
  const h16* Arow = A + (size_t)(tm * 16 + l15) * lda;
  const int bn = tn * 16 + l15;
  const h16* Bcol = Bt + (size_t)bn * Kpad;
  for (int k0 = 0; k0 < Kpad; k0 += 32) {
    // A fragment (16x32): lanes 0-15 hold K {0..7,16..23}, lanes 16-31 {8..15,24..31}
    const v8h* pa = (const v8h*)(Arow + k0 + sel * 8);
    v8h alo = pa[0];                           // k0+sel*8 .. +7
    v8h ahi = pa[2];                           // +16 halfs
    // B fragment (32x16): col = lane&15, lanes 0-15 K {0..7,16..23}, lanes 16-31 {8..15,24..31}
    const v8h* pb = (const v8h*)(Bcol + k0 + sel * 8);
    v8h b0 = pb[0], b1 = pb[2];
    v16h a, b;
#pragma unroll
    for (int e = 0; e < 8; ++e) {
      a[e] = alo[e]; a[e + 8] = ahi[e];
      b[e] = b0[e];  b[e + 8] = b1[e];
    }
    acc = __builtin_amdgcn_wmma_f32_16x16x32_f16(false, a, false, b, (short)0, acc,
                                                 false, false);
  }
  const float bv = bias ? bias[bn] : 0.f;
#pragma unroll
  for (int r = 0; r < 8; ++r) {
    int m = tm * 16 + r + sel * 8;             // C/D layout: vgpr r -> row r (+8 hi lanes)
    float v = acc[r] + bv;
    if (ACT == 1) v = fmaxf(v, 0.f);
    else if (ACT == 2) v = 0.5f * v * (1.f + erff(v * 0.70710678118f));
    size_t o = (size_t)m * N + bn;
    if (outH) outH[o] = (h16)v;
    if (outF) outF[o] = v;
  }
}

// Pack fp32 weight [Kin,N] into fp16 TRANSPOSED [N,Kpad] (zero pad k >= Kin).
__global__ void pack_weight(const float* __restrict__ w, h16* __restrict__ dst,
                            int Kin, int Kpad, int N)
{
  int i = blockIdx.x * 256 + threadIdx.x;
  if (i >= Kpad * N) return;
  int n = i / Kpad, k = i % Kpad;
  dst[i] = (h16)((k < Kin) ? w[k * N + n] : 0.f);
}

__global__ void zero_f32(float* p, int n)
{
  int i = blockIdx.x * 256 + threadIdx.x;
  if (i < n) p[i] = 0.f;
}

// pp[b,n] = float4(x,y,z,|p|^2) with p = P + offsets (offsets in [B,3,N] layout)
__global__ void build_pp(const float* __restrict__ P, const float* __restrict__ off,
                         float4* __restrict__ pp)
{
  int i = blockIdx.x * 256 + threadIdx.x;
  if (i >= NB * NPTS) return;
  int b = i / NPTS, n = i % NPTS;
  const float* Pb = P + (size_t)b * 3 * NPTS;
  const float* Ob = off + (size_t)b * 3 * NPTS;
  float x = Pb[n] + Ob[n];
  float y = Pb[NPTS + n] + Ob[NPTS + n];
  float z = Pb[2 * NPTS + n] + Ob[2 * NPTS + n];
  pp[i] = make_float4(x, y, z, x * x + y * y + z * z);
}

// ---------------------------------------------------------------------------
// Self-KNN: one thread per query, points streamed through LDS in 2048-pt
// (32KB) tiles via async global->LDS b128 copies. Register-resident sorted
// top-KS list with guarded unrolled insertion.
// ---------------------------------------------------------------------------
template<int KS>
__launch_bounds__(256)
__global__ void knn_kernel(const float4* __restrict__ pp,
                           int* __restrict__ idxOut, float* __restrict__ d0Out)
{
  __shared__ float4 tile[2048];
  const int b = blockIdx.x / (NPTS / 256);
  const int n = (blockIdx.x % (NPTS / 256)) * 256 + threadIdx.x;
  const float4* ppb = pp + (size_t)b * NPTS;
  const float4 q = ppb[n];
  float bd[KS]; int bi[KS];
#pragma unroll
  for (int s = 0; s < KS; ++s) { bd[s] = 3.4e38f; bi[s] = 0; }

  const uint32_t lbase = (uint32_t)(uintptr_t)&tile[0];
  const uint64_t gbase = (uint64_t)(uintptr_t)ppb;
  for (int t0 = 0; t0 < NPTS; t0 += 2048) {
#pragma unroll
    for (int i = 0; i < 8; ++i) {
      uint32_t j = (uint32_t)threadIdx.x + 256u * i;
      uint32_t voff = (uint32_t)(t0 + j) * 16u;
      uint32_t laddr = lbase + j * 16u;
      asm volatile("global_load_async_to_lds_b128 %0, %1, %2 offset:0"
                   :: "v"(laddr), "v"(voff), "s"(gbase) : "memory");
    }
    asm volatile("s_wait_asynccnt 0" ::: "memory");
    __syncthreads();
    for (int j = 0; j < 2048; ++j) {
      float4 c = tile[j];
      float d = q.w + c.w - 2.f * (q.x * c.x + q.y * c.y + q.z * c.z);
      if (d < bd[KS - 1]) {
        float cd = d; int ci = t0 + j;
#pragma unroll
        for (int s = 0; s < KS; ++s) {
          bool sw = cd < bd[s];
          float td = bd[s]; int ti = bi[s];
          bd[s] = sw ? cd : td; bi[s] = sw ? ci : ti;
          cd = sw ? td : cd;    ci = sw ? ti : ci;
        }
      }
    }
    __syncthreads();
  }
  if (KS == KNN) {
    int* o = idxOut + ((size_t)b * NPTS + n) * KNN;
#pragma unroll
    for (int s = 0; s < KS; ++s) o[s] = bi[s];
  } else {
    d0Out[(size_t)b * NPTS + n] = sqrtf(fmaxf(bd[KS - 1], 1e-12f));
  }
}

// init feature MLP: geom=[d0,d0^2,0,0,0,0] -> relu(W1)+LN -> W2 -> h[64]
__global__ void init_feat(const float* __restrict__ d0,
                          const float* __restrict__ w1, const float* __restrict__ b1,
                          const float* __restrict__ lng, const float* __restrict__ lnb,
                          const float* __restrict__ w2, const float* __restrict__ b2,
                          float* __restrict__ h)
{
  int i = blockIdx.x * 256 + threadIdx.x;
  if (i >= NB * NPTS) return;
  float d = d0[i], g0 = d, g1 = d * d;
  float a[64]; float mean = 0.f;
  for (int j = 0; j < 64; ++j) {
    float v = g0 * w1[j] + g1 * w1[64 + j] + b1[j];
    v = fmaxf(v, 0.f); a[j] = v; mean += v;
  }
  mean *= (1.f / 64.f);
  float var = 0.f;
  for (int j = 0; j < 64; ++j) { float t = a[j] - mean; var += t * t; }
  var *= (1.f / 64.f);
  float inv = rsqrtf(var + 1e-5f);
  for (int j = 0; j < 64; ++j) a[j] = (a[j] - mean) * inv * lng[j] + lnb[j];
  float* hr = h + (size_t)i * 64;
  for (int j = 0; j < 64; ++j) {
    float s = b2[j];
    for (int c = 0; c < 64; ++c) s += a[c] * w2[c * 64 + j];
    hr[j] = s;
  }
}

// tri-net + stats: per point, scan 32 neighbors, 6->32->1 MLP, sigmoid stats.
__global__ void tri_stats(const float4* __restrict__ pp, const float* __restrict__ off,
                          const int* __restrict__ idx,
                          const float* __restrict__ tw1, const float* __restrict__ tb1,
                          const float* __restrict__ tw2, const float* __restrict__ tb2,
                          float* __restrict__ q4)
{
  int i = blockIdx.x * 256 + threadIdx.x;
  if (i >= NB * NPTS) return;
  int b = i / NPTS, n = i % NPTS;
  const float4* ppb = pp + (size_t)b * NPTS;
  const float* ob = off + (size_t)b * 3 * NPTS;
  float4 q = ppb[n];
  float ox = ob[n], oy = ob[NPTS + n], oz = ob[2 * NPTS + n];
  const int* ix = idx + (size_t)i * KNN;
  float pmax = 0.f, psum = 0.f, esum = 0.f, m1 = -3.4e38f, m2 = -3.4e38f;
  for (int k = 0; k < KNN; ++k) {
    int nb = ix[k];
    float4 c = ppb[nb];
    float x0 = c.x - q.x, x1 = c.y - q.y, x2 = c.z - q.z;
    float x3 = ob[nb] - ox, x4 = ob[NPTS + nb] - oy, x5 = ob[2 * NPTS + nb] - oz;
    float logit = tb2[0];
    for (int j = 0; j < 32; ++j) {
      float v = x0 * tw1[j] + x1 * tw1[32 + j] + x2 * tw1[64 + j]
              + x3 * tw1[96 + j] + x4 * tw1[128 + j] + x5 * tw1[160 + j] + tb1[j];
      logit += fmaxf(v, 0.f) * tw2[j];
    }
    float p = 1.f / (1.f + expf(-logit));
    pmax = fmaxf(pmax, p); psum += p;
    esum += -(p * logf(p + 1e-6f) + (1.f - p) * logf(1.f - p + 1e-6f));
    if (logit > m1) { m2 = m1; m1 = logit; } else if (logit > m2) m2 = logit;
  }
  float* o = q4 + (size_t)i * 4;
  o[0] = pmax; o[1] = psum * (1.f / KNN); o[2] = esum * (1.f / KNN); o[3] = m1 - m2;
}

// concat [h(64), q4(4), dP(3)] -> fp16 [M,96] (zero padded)
__global__ void build_xin(const float* __restrict__ h, const float* __restrict__ q4,
                          const float* __restrict__ off, h16* __restrict__ xin)
{
  int i = blockIdx.x * 256 + threadIdx.x;
  if (i >= NB * NPTS) return;
  int b = i / NPTS, n = i % NPTS;
  const float* ob = off + (size_t)b * 3 * NPTS;
  h16* o = xin + (size_t)i * 96;
  const float* hr = h + (size_t)i * 64;
  for (int j = 0; j < 64; ++j) o[j] = (h16)hr[j];
  const float* qr = q4 + (size_t)i * 4;
  for (int j = 0; j < 4; ++j) o[64 + j] = (h16)qr[j];
  o[68] = (h16)ob[n]; o[69] = (h16)ob[NPTS + n]; o[70] = (h16)ob[2 * NPTS + n];
  for (int j = 71; j < 96; ++j) o[j] = (h16)0.f;
}

// LayerNorm over 64 channels of (a [+ b]); writes f32 and/or f16.
__global__ void ln_kernel(const float* __restrict__ a, const float* __restrict__ b,
                          const float* __restrict__ g, const float* __restrict__ bt,
                          float* __restrict__ y32, h16* __restrict__ y16, int M)
{
  int i = blockIdx.x * 256 + threadIdx.x;
  if (i >= M) return;
  const float* ar = a + (size_t)i * 64;
  const float* br = b ? b + (size_t)i * 64 : nullptr;
  float v[64]; float mean = 0.f;
  for (int j = 0; j < 64; ++j) { float t = ar[j] + (br ? br[j] : 0.f); v[j] = t; mean += t; }
  mean *= (1.f / 64.f);
  float var = 0.f;
  for (int j = 0; j < 64; ++j) { float t = v[j] - mean; var += t * t; }
  var *= (1.f / 64.f);
  float inv = rsqrtf(var + 1e-5f);
  for (int j = 0; j < 64; ++j) {
    float t = (v[j] - mean) * inv * g[j] + bt[j];
    if (y32) y32[(size_t)i * 64 + j] = t;
    if (y16) y16[(size_t)i * 64 + j] = (h16)t;
  }
}

// geometry rows for the pos-encoding MLP: [relx,rely,relz,dist,0...] fp16 [Mrows,32]
__global__ void build_geom(const float4* __restrict__ pp, const int* __restrict__ idx,
                           int base, h16* __restrict__ geom, int Mrows)
{
  int m = blockIdx.x * 256 + threadIdx.x;
  if (m >= Mrows) return;
  int nl = m >> 5, k = m & 31;
  int gp = base + nl; int b = gp / NPTS, n = gp % NPTS;
  const float4* ppb = pp + (size_t)b * NPTS;
  int nb = idx[((size_t)b * NPTS + n) * KNN + k];
  float4 q = ppb[n], c = ppb[nb];
  float rx = c.x - q.x, ry = c.y - q.y, rz = c.z - q.z;
  float dd = sqrtf(rx * rx + ry * ry + rz * rz);
  h16* o = geom + (size_t)m * 32;
  o[0] = (h16)rx; o[1] = (h16)ry; o[2] = (h16)rz; o[3] = (h16)dd;
  for (int j = 4; j < 32; ++j) o[j] = (h16)0.f;
}

// relation = Q - K(nbr) + pe, fp16 [Mrows,64]
__global__ void build_rel(const h16* __restrict__ Q16, const h16* __restrict__ K16,
                          const h16* __restrict__ pe, const int* __restrict__ idx,
                          int base, h16* __restrict__ rel, int Mrows)
{
  int t = blockIdx.x * 256 + threadIdx.x;
  if (t >= Mrows * 64) return;
  int m = t >> 6, c = t & 63;
  int nl = m >> 5, k = m & 31;
  int gp = base + nl; int b = gp / NPTS, n = gp % NPTS;
  int nb = idx[((size_t)b * NPTS + n) * KNN + k];
  float v = (float)Q16[(size_t)gp * 64 + c]
          - (float)K16[((size_t)b * NPTS + nb) * 64 + c]
          + (float)pe[(size_t)m * 64 + c];
  rel[(size_t)m * 64 + c] = (h16)v;
}

// softmax over neighbors + aggregate w*(V+pe): one wave per (point,head).
__launch_bounds__(256)
__global__ void attn_agg(const h16* __restrict__ amh, const float* __restrict__ amw2,
                         const float* __restrict__ amb2, const h16* __restrict__ V16,
                         const h16* __restrict__ pe, const int* __restrict__ idx,
                         int base, h16* __restrict__ agg, int npts)
{
  int lane = threadIdx.x & 31, w = threadIdx.x >> 5;
  int pair = blockIdx.x * 8 + w;
  if (pair >= npts * 4) return;
  int nl = pair >> 2, head = pair & 3;
  int gp = base + nl; int b = gp / NPTS, n = gp % NPTS;
  int k = lane;
  int m = nl * KNN + k;
  const h16* ar = amh + ((size_t)m * 4 + head) * 16;
  float logit = amb2[0];
#pragma unroll
  for (int j = 0; j < 16; ++j) logit += (float)ar[j] * amw2[j];
  float mx = logit;
  for (int s = 16; s > 0; s >>= 1) mx = fmaxf(mx, __shfl_xor(mx, s, 32));
  float e = expf(logit - mx), sum = e;
  for (int s = 16; s > 0; s >>= 1) sum += __shfl_xor(sum, s, 32);
  float wk = e / sum;
  int nb = idx[((size_t)b * NPTS + n) * KNN + k];
#pragma unroll
  for (int c16 = 0; c16 < 16; ++c16) {
    int c = head * 16 + c16;
    float v = wk * ((float)V16[((size_t)b * NPTS + nb) * 64 + c]
                    + (float)pe[(size_t)m * 64 + c]);
    for (int s = 16; s > 0; s >>= 1) v += __shfl_xor(v, s, 32);
    if (lane == 0) agg[(size_t)gp * 64 + c] = (h16)v;
  }
}

// offsets += step_scale * (hn @ off_w + off_b), written in [B,3,N] layout
__global__ void update_off(const float* __restrict__ hn, const float* __restrict__ offw,
                           const float* __restrict__ offb, const float* __restrict__ step,
                           float* __restrict__ out)
{
  int i = blockIdx.x * 256 + threadIdx.x;
  if (i >= NB * NPTS) return;
  int b = i / NPTS, n = i % NPTS;
  const float* hr = hn + (size_t)i * 64;
  float ss = step[0];
  for (int j = 0; j < 3; ++j) {
    float s = offb[j];
    for (int c = 0; c < 64; ++c) s += hr[c] * offw[c * 3 + j];
    out[(size_t)b * 3 * NPTS + (size_t)j * NPTS + n] += s * ss;
  }
}

// ---------------------------------------------------------------------------
extern "C" void kernel_launch(void* const* d_in, const int* in_sizes, int n_in,
                              void* d_out, int out_size, void* d_ws, size_t ws_size,
                              hipStream_t stream)
{
  (void)in_sizes; (void)n_in; (void)ws_size;
  enum { I_P = 0, I_AM_B1, I_AM_B2, I_AM_W1, I_AM_W2, I_FFN_B1, I_FFN_B2, I_FFN_W1,
         I_FFN_W2, I_INIT_B1, I_INIT_B2, I_INIT_LNB, I_INIT_LNG, I_INIT_W1, I_INIT_W2,
         I_K_W, I_LN1_B, I_LN1_G, I_LN2_B, I_LN2_G, I_LN3_B, I_LN3_G, I_OFF_B, I_OFF_W,
         I_OUT_B, I_OUT_W, I_POS_B1, I_POS_B2, I_POS_W1, I_POS_W2, I_PROJ_B, I_PROJ_W,
         I_Q_W, I_STEP, I_TRI_B1, I_TRI_B2, I_TRI_W1, I_TRI_W2, I_V_W };
  auto F = [&](int i) { return (const float*)d_in[i]; };
  float* out = (float*)d_out;
  const int M1 = NB * NPTS;
  const int M2c = CHP * KNN;

  char* wp = (char*)d_ws;
  auto alloc = [&](size_t bytes) -> void* {
    void* r = (void*)wp; wp += (bytes + 255) & ~(size_t)255; return r;
  };
  float4* pp   = (float4*)alloc((size_t)M1 * sizeof(float4));
  float*  d0   = (float*)alloc((size_t)M1 * 4);
  float*  hbuf = (float*)alloc((size_t)M1 * 64 * 4);
  int*    idx  = (int*)alloc((size_t)M1 * KNN * 4);
  float*  q4   = (float*)alloc((size_t)M1 * 4 * 4);
  h16*    xin  = (h16*)alloc((size_t)M1 * 96 * 2);
  float*  ta32 = (float*)alloc((size_t)M1 * 64 * 4);
  float*  tb32 = (float*)alloc((size_t)M1 * 64 * 4);
  float*  x32  = (float*)alloc((size_t)M1 * 64 * 4);
  h16*    x16  = (h16*)alloc((size_t)M1 * 64 * 2);
  h16*    Q16  = (h16*)alloc((size_t)M1 * 64 * 2);
  h16*    K16  = (h16*)alloc((size_t)M1 * 64 * 2);
  h16*    V16  = (h16*)alloc((size_t)M1 * 64 * 2);
  h16*    agg  = (h16*)alloc((size_t)M1 * 64 * 2);
  h16*    ffh  = (h16*)alloc((size_t)M1 * 128 * 2);
  h16*    geom = (h16*)alloc((size_t)M2c * 32 * 2);
  h16*    hpos = (h16*)alloc((size_t)M2c * 64 * 2);   // reused as amh [M2c*4,16]
  h16*    pe   = (h16*)alloc((size_t)M2c * 64 * 2);
  h16*    rel  = (h16*)alloc((size_t)M2c * 64 * 2);
  h16* Wproj = (h16*)alloc(96 * 64 * 2);
  h16* Wq    = (h16*)alloc(64 * 64 * 2);
  h16* Wk    = (h16*)alloc(64 * 64 * 2);
  h16* Wv    = (h16*)alloc(64 * 64 * 2);
  h16* Wpos1 = (h16*)alloc(32 * 64 * 2);
  h16* Wpos2 = (h16*)alloc(64 * 64 * 2);
  h16* Wam1  = (h16*)alloc(32 * 16 * 2);
  h16* Wout  = (h16*)alloc(64 * 64 * 2);
  h16* Wffn1 = (h16*)alloc(64 * 128 * 2);
  h16* Wffn2 = (h16*)alloc(128 * 64 * 2);

  auto packw = [&](const float* w, h16* dst, int Kin, int Kpad, int N) {
    pack_weight<<<(Kpad * N + 255) / 256, 256, 0, stream>>>(w, dst, Kin, Kpad, N);
  };
  auto gemm = [&](int act, const h16* A, int M, int Kpad, int lda,
                  const h16* Bt, int N, const float* bias, h16* oh, float* of) {
    int tiles = (M / 16) * (N / 16);
    dim3 g((unsigned)((tiles + 7) / 8));
    if (act == 0)      gemm_wmma<0><<<g, 256, 0, stream>>>(A, Kpad, lda, Bt, N, tiles, bias, oh, of);
    else if (act == 1) gemm_wmma<1><<<g, 256, 0, stream>>>(A, Kpad, lda, Bt, N, tiles, bias, oh, of);
    else               gemm_wmma<2><<<g, 256, 0, stream>>>(A, Kpad, lda, Bt, N, tiles, bias, oh, of);
  };
  const int g256 = (M1 + 255) / 256;

  zero_f32<<<(out_size + 255) / 256, 256, 0, stream>>>(out, out_size);

  packw(F(I_PROJ_W), Wproj, 71, 96, 64);
  packw(F(I_Q_W), Wq, 64, 64, 64);
  packw(F(I_K_W), Wk, 64, 64, 64);
  packw(F(I_V_W), Wv, 64, 64, 64);
  packw(F(I_POS_W1), Wpos1, 4, 32, 64);
  packw(F(I_POS_W2), Wpos2, 64, 64, 64);
  packw(F(I_AM_W1), Wam1, 16, 32, 16);
  packw(F(I_OUT_W), Wout, 64, 64, 64);
  packw(F(I_FFN_W1), Wffn1, 64, 64, 128);
  packw(F(I_FFN_W2), Wffn2, 128, 128, 64);

  // init features from nearest-neighbor distance
  build_pp<<<g256, 256, 0, stream>>>(F(I_P), out, pp);
  knn_kernel<2><<<NB * (NPTS / 256), 256, 0, stream>>>(pp, nullptr, d0);
  init_feat<<<g256, 256, 0, stream>>>(d0, F(I_INIT_W1), F(I_INIT_B1), F(I_INIT_LNG),
                                      F(I_INIT_LNB), F(I_INIT_W2), F(I_INIT_B2), hbuf);

  for (int t = 0; t < 2; ++t) {
    build_pp<<<g256, 256, 0, stream>>>(F(I_P), out, pp);
    knn_kernel<KNN><<<NB * (NPTS / 256), 256, 0, stream>>>(pp, idx, nullptr);
    tri_stats<<<g256, 256, 0, stream>>>(pp, out, idx, F(I_TRI_W1), F(I_TRI_B1),
                                        F(I_TRI_W2), F(I_TRI_B2), q4);
    build_xin<<<g256, 256, 0, stream>>>(hbuf, q4, out, xin);

    gemm(0, xin, M1, 96, 96, Wproj, 64, F(I_PROJ_B), nullptr, ta32);
    ln_kernel<<<g256, 256, 0, stream>>>(ta32, nullptr, F(I_LN1_G), F(I_LN1_B), x32, x16, M1);
    gemm(0, x16, M1, 64, 64, Wq, 64, nullptr, Q16, nullptr);
    gemm(0, x16, M1, 64, 64, Wk, 64, nullptr, K16, nullptr);
    gemm(0, x16, M1, 64, 64, Wv, 64, nullptr, V16, nullptr);

    for (int base = 0; base < M1; base += CHP) {
      build_geom<<<(M2c + 255) / 256, 256, 0, stream>>>(pp, idx, base, geom, M2c);
      gemm(1, geom, M2c, 32, 32, Wpos1, 64, F(I_POS_B1), hpos, nullptr);
      gemm(0, hpos, M2c, 64, 64, Wpos2, 64, F(I_POS_B2), pe, nullptr);
      build_rel<<<(M2c * 64 + 255) / 256, 256, 0, stream>>>(Q16, K16, pe, idx, base, rel, M2c);
      // rel viewed as [M2c*4, 16] rows; Wam1 rows 16..31 are zero so the
      // unconditional 32-wide A reads contribute nothing beyond k=16.
      gemm(1, rel, M2c * 4, 32, 16, Wam1, 16, F(I_AM_B1), hpos /*amh*/, nullptr);
      attn_agg<<<(CHP * 4 + 7) / 8, 256, 0, stream>>>(hpos, F(I_AM_W2), F(I_AM_B2),
                                                      V16, pe, idx, base, agg, CHP);
    }
    gemm(0, agg, M1, 64, 64, Wout, 64, F(I_OUT_B), nullptr, ta32);
    ln_kernel<<<g256, 256, 0, stream>>>(x32, ta32, F(I_LN2_G), F(I_LN2_B), tb32, x16, M1);
    gemm(2, x16, M1, 64, 64, Wffn1, 128, F(I_FFN_B1), ffh, nullptr);
    gemm(0, ffh, M1, 128, 128, Wffn2, 64, F(I_FFN_B2), nullptr, ta32);
    ln_kernel<<<g256, 256, 0, stream>>>(tb32, ta32, F(I_LN3_G), F(I_LN3_B), hbuf, nullptr, M1);
    update_off<<<g256, 256, 0, stream>>>(hbuf, F(I_OFF_W), F(I_OFF_B), F(I_STEP), out);
  }
}
